// SPP_73418170958019
// MI455X (gfx1250) — matrix operand order repaired and checked by reading, not compiled, not run on hardware
//
#include <hip/hip_runtime.h>
#include <hip/hip_fp16.h>

typedef __attribute__((ext_vector_type(16))) _Float16 v16h;
typedef __attribute__((ext_vector_type(8)))  float    v8f;
typedef __attribute__((ext_vector_type(2)))  __fp16   pk2h;   // matches cvt_pkrtz return type

// Problem constants (from reference): N=1M pts, C=64, OUT=96, GRID=128, B=4, kernels {2,4,8}
static constexpr int C_IN   = 64;
static constexpr int C_OUT  = 96;
static constexpr int SEG_B1 = 4;                    // after 4 global segs
static constexpr int SEG_B2 = SEG_B1 + (1 << 20);   // + 4*64^3
static constexpr int SEG_B3 = SEG_B2 + (1 << 17);   // + 4*32^3
static constexpr int S_TOT  = SEG_B3 + (1 << 14);   // + 4*16^3 = 1,196,036 rows
// ws layout (floats): sums[S_TOT*64] | cnts[S_TOT] | Wh (f16, 96*256) => ~311 MB

__device__ __forceinline__ unsigned pk16(float a, float b) {
    union { pk2h h; unsigned u; } u;
    u.h = __builtin_amdgcn_cvt_pkrtz(a, b);   // v_cvt_pk_rtz_f16_f32
    return u.u;
}

// ---------------- kernel 1: zero sums + counts ----------------
__global__ void k_zero(uint4* __restrict__ ws, long n4) {
    long i = (long)blockIdx.x * blockDim.x + threadIdx.x;
    long st = (long)gridDim.x * blockDim.x;
    uint4 z = {0u, 0u, 0u, 0u};
    for (; i < n4; i += st) ws[i] = z;
}

// ---------------- kernel 2: W f32 -> f16 ----------------
__global__ void k_wconv(const float* __restrict__ W, unsigned* __restrict__ Wh) {
    int i = blockIdx.x * blockDim.x + threadIdx.x;    // one packed pair per thread
    if (i < C_OUT * 256 / 2) {
        float2 f = ((const float2*)W)[i];
        Wh[i] = pk16(f.x, f.y);
    }
}

// ---------------- kernel 3: scatter segment sums ----------------
__global__ __launch_bounds__(256) void k_scatter(
    const float* __restrict__ feats, const int* __restrict__ coords,
    const int* __restrict__ bidx, float* __restrict__ sums,
    float* __restrict__ cnts, int N)
{
    __shared__ float s0[4 * C_IN];   // per-block partial for global level (4 batches x 64 ch)
    __shared__ float s0c[4];
    int t = threadIdx.x;
    if (t < 4 * C_IN) s0[t] = 0.f;
    if (t < 4) s0c[t] = 0.f;
    __syncthreads();

    int n = blockIdx.x * 256 + t;
    if (n < N) {
        int b  = bidx[n];
        int c0 = coords[3 * n], c1 = coords[3 * n + 1], c2 = coords[3 * n + 2];
        int s1 = SEG_B1 + ((((b << 6) | (c0 >> 1)) << 6 | (c1 >> 1)) << 6 | (c2 >> 1));
        int s2 = SEG_B2 + ((((b << 5) | (c0 >> 2)) << 5 | (c1 >> 2)) << 5 | (c2 >> 2));
        int s3 = SEG_B3 + ((((b << 4) | (c0 >> 3)) << 4 | (c1 >> 3)) << 4 | (c2 >> 3));
        const float4* f4 = (const float4*)(feats + (long)n * C_IN);
        float* p0 = s0 + b * C_IN;
        float* p1 = sums + (long)s1 * C_IN;
        float* p2 = sums + (long)s2 * C_IN;
        float* p3 = sums + (long)s3 * C_IN;
        #pragma unroll 4
        for (int g = 0; g < 16; ++g) {
            float4 v = f4[g];
            int c = g * 4;
            atomicAdd(p0 + c + 0, v.x); atomicAdd(p0 + c + 1, v.y);   // ds_add_f32
            atomicAdd(p0 + c + 2, v.z); atomicAdd(p0 + c + 3, v.w);
            atomicAdd(p1 + c + 0, v.x); atomicAdd(p1 + c + 1, v.y);   // global_atomic_add_f32
            atomicAdd(p1 + c + 2, v.z); atomicAdd(p1 + c + 3, v.w);
            atomicAdd(p2 + c + 0, v.x); atomicAdd(p2 + c + 1, v.y);
            atomicAdd(p2 + c + 2, v.z); atomicAdd(p2 + c + 3, v.w);
            atomicAdd(p3 + c + 0, v.x); atomicAdd(p3 + c + 1, v.y);
            atomicAdd(p3 + c + 2, v.z); atomicAdd(p3 + c + 3, v.w);
        }
        atomicAdd(&cnts[s1], 1.f);
        atomicAdd(&cnts[s2], 1.f);
        atomicAdd(&cnts[s3], 1.f);
        atomicAdd(&s0c[b], 1.f);
    }
    __syncthreads();
    if (t < 4 * C_IN) atomicAdd(&sums[t], s0[t]);   // global-level rows are 0..3
    if (t < 4) atomicAdd(&cnts[t], s0c[t]);
}

// ---------------- kernel 4: sums -> f16 means (in place, row-packed) ----------------
__global__ void k_means(float* __restrict__ sums, const float* __restrict__ cnts) {
    int r = blockIdx.x * blockDim.x + threadIdx.x;
    if (r >= S_TOT) return;
    float inv = 1.f / fmaxf(cnts[r], 1.f);
    float4* row = (float4*)(sums + (long)r * C_IN);
    float v[64];
    #pragma unroll
    for (int g = 0; g < 16; ++g) {        // read whole row first (aliasing-safe)
        float4 a = row[g];
        v[4*g+0] = a.x * inv; v[4*g+1] = a.y * inv;
        v[4*g+2] = a.z * inv; v[4*g+3] = a.w * inv;
    }
    uint4* dst = (uint4*)row;             // f16 means occupy first 128B of the 256B row
    #pragma unroll
    for (int g = 0; g < 8; ++g) {
        uint4 o;
        o.x = pk16(v[8*g+0], v[8*g+1]);
        o.y = pk16(v[8*g+2], v[8*g+3]);
        o.z = pk16(v[8*g+4], v[8*g+5]);
        o.w = pk16(v[8*g+6], v[8*g+7]);
        dst[g] = o;
    }
}

// ---------------- kernel 5: gather + WMMA GEMM (+bias) ----------------
// Tile: 128 points x 96 outputs. 8 waves; wave w owns rows [16w,16w+16), all 6 N-tiles.
#define TILE_M 128
#define ROWW   132   // padded row width in u32 words (264 halves) to dodge bank conflicts

__global__ __launch_bounds__(256) void k_gemm(
    const unsigned* __restrict__ means,   // row r: f16 means in words [r*64, r*64+32)
    const unsigned* __restrict__ Wh,      // [96][128] u32 words (f16 pairs, row-major K)
    const float* __restrict__ bias,
    const int* __restrict__ coords, const int* __restrict__ bidx,
    float* __restrict__ out, int N)
{
    __shared__ __align__(16) unsigned a_lds[TILE_M * ROWW];
    __shared__ __align__(16) unsigned w_lds[C_OUT * ROWW];
    __shared__ float b_lds[C_OUT];
    int t = threadIdx.x;
    int pbase = blockIdx.x * TILE_M;

    // stage W + bias into LDS
    for (int i = t; i < C_OUT * 128; i += 256)
        w_lds[(i >> 7) * ROWW + (i & 127)] = Wh[i];
    if (t < C_OUT) b_lds[t] = bias[t];

    // gather: 4 level-rows per point, 512 tasks
    for (int task = t; task < TILE_M * 4; task += 256) {
        int p = task >> 2, lvl = task & 3;
        int n = pbase + p;
        if (n >= N) n = N - 1;
        int b  = bidx[n];
        int c0 = coords[3 * n], c1 = coords[3 * n + 1], c2 = coords[3 * n + 2];
        int r;
        if      (lvl == 0) r = b;
        else if (lvl == 1) r = SEG_B1 + ((((b << 6) | (c0 >> 1)) << 6 | (c1 >> 1)) << 6 | (c2 >> 1));
        else if (lvl == 2) r = SEG_B2 + ((((b << 5) | (c0 >> 2)) << 5 | (c1 >> 2)) << 5 | (c2 >> 2));
        else               r = SEG_B3 + ((((b << 4) | (c0 >> 3)) << 4 | (c1 >> 3)) << 4 | (c2 >> 3));
        const uint4* src = (const uint4*)(means + (long)r * 64);   // 8 x uint4 = 64 f16
        uint4* dst = (uint4*)&a_lds[p * ROWW + lvl * 32];
        #pragma unroll
        for (int i = 0; i < 8; ++i) dst[i] = src[i];
    }
    __syncthreads();

    int wave = t >> 5, lane = t & 31;
    int lm = lane & 15, lh = lane >> 4;
    int mrow = wave * 16;

    v8f acc[6];
    #pragma unroll
    for (int nt = 0; nt < 6; ++nt) acc[nt] = (v8f){0.f,0.f,0.f,0.f,0.f,0.f,0.f,0.f};

    const unsigned* arow = &a_lds[(mrow + lm) * ROWW];
    #pragma unroll
    for (int kk = 0; kk < 8; ++kk) {                 // 8 chunks of K=32 halves (16 words)
        int kw = kk * 16;
        union { v16h h; uint4 q[2]; } A;             // 16-bit A layout: {0..3,8..11}+4*hi
        A.q[0] = *(const uint4*)&arow[kw + 4 * lh];
        A.q[1] = *(const uint4*)&arow[kw + 8 + 4 * lh];
        #pragma unroll
        for (int nt = 0; nt < 6; ++nt) {
            union { v16h h; uint4 q[2]; } Bf;        // 16-bit B layout: {0..3,8..11}+4*hi (same as A)
            const unsigned* wrow = &w_lds[(nt * 16 + lm) * ROWW];
            Bf.q[0] = *(const uint4*)&wrow[kw + 4 * lh];
            Bf.q[1] = *(const uint4*)&wrow[kw + 8 + 4 * lh];
            acc[nt] = __builtin_amdgcn_wmma_f32_16x16x32_f16(
                false, A.h, false, Bf.h, (short)0, acc[nt], false, false);
        }
    }

    #pragma unroll
    for (int nt = 0; nt < 6; ++nt) {
        int col = nt * 16 + lm;
        float bv = b_lds[col];
        #pragma unroll
        for (int v = 0; v < 8; ++v) {                // D: M = 8*hi + v, N = lm
            int m = mrow + lh * 8 + v;
            if (pbase + m < N)
                out[(long)(pbase + m) * C_OUT + col] = acc[nt][v] + bv;
        }
    }
}

extern "C" void kernel_launch(void* const* d_in, const int* in_sizes, int n_in,
                              void* d_out, int out_size, void* d_ws, size_t ws_size,
                              hipStream_t stream) {
    const float* feats  = (const float*)d_in[0];
    const int*   coords = (const int*)d_in[1];
    const int*   bix    = (const int*)d_in[2];
    const float* W      = (const float*)d_in[3];
    const float* bias   = (const float*)d_in[4];
    int N = in_sizes[0] / C_IN;

    float*    sums = (float*)d_ws;
    float*    cnts = sums + (size_t)S_TOT * C_IN;
    unsigned* Wh   = (unsigned*)(cnts + S_TOT);

    long n4 = ((long)S_TOT * (C_IN + 1)) / 4;   // sums + counts, in uint4
    k_zero   <<<4096, 256, 0, stream>>>((uint4*)d_ws, n4);
    k_wconv  <<<(C_OUT * 128 + 255) / 256, 256, 0, stream>>>(W, Wh);
    k_scatter<<<(N + 255) / 256, 256, 0, stream>>>(feats, coords, bix, sums, cnts, N);
    k_means  <<<(S_TOT + 255) / 256, 256, 0, stream>>>(sums, cnts);
    k_gemm   <<<(N + TILE_M - 1) / TILE_M, 256, 0, stream>>>(
        (const unsigned*)sums, Wh, bias, coords, bix, (float*)d_out, N);
}
